// InteractionBlock_64819646431979
// MI455X (gfx1250) — matrix operand order repaired and checked by reading, not compiled, not run on hardware
//
#include <hip/hip_runtime.h>
#include <hip/hip_bf16.h>

#define HIDDEN 128
#define NUM_RBF 50

typedef __bf16          v16bf __attribute__((ext_vector_type(16)));
typedef float           v8f   __attribute__((ext_vector_type(8)));
typedef unsigned short  us8   __attribute__((ext_vector_type(8)));

// ---------- helpers -------------------------------------------------------

// Native hardware convert (v_cvt_*bf16_f32, RNE) instead of 3-op bit twiddling.
static __device__ __forceinline__ unsigned short f32_to_bf16_bits(float f) {
  __bf16 b = (__bf16)f;
  return __builtin_bit_cast(unsigned short, b);
}

static __device__ __forceinline__ float fast_tanh(float x) {
  // tanh(x) = 1 - 2/(e^{2x}+1); v_exp_f32 + v_rcp_f32, saturates correctly at +-inf
  float e = __expf(2.0f * x);
  return 1.0f - 2.0f * __builtin_amdgcn_rcpf(e + 1.0f);
}

union FragU { v16bf v; unsigned short u[16]; us8 h2[2]; };

static __device__ __forceinline__ v8f wmma_bf16(v16bf a, v16bf b, v8f c) {
  // v_wmma_f32_16x16x32_bf16 : D = A(16x32 bf16) * B(32x16 bf16) + C(f32)
  return __builtin_amdgcn_wmma_f32_16x16x32_bf16(false, a, false, b, (short)0, c,
                                                 false, false);
}

// B fragment: packed contiguously per lane (16 bf16 = 32B), two b128 loads.
static __device__ __forceinline__ v16bf load_bfrag(const unsigned short* p) {
  FragU f;
  f.h2[0] = *(const us8*)(p);
  f.h2[1] = *(const us8*)(p + 8);
  return f.v;
}

// A fragment from an LDS 16x128 bf16 row-major tile.
// lane l: row m=l&15; K chunks at k0+8*(l>>4) and k0+16+8*(l>>4).
static __device__ __forceinline__ v16bf load_afrag_lds(const unsigned short* tile,
                                                       int lane, int k0) {
  int m = lane & 15, hi = lane >> 4;
  const unsigned short* p = tile + m * HIDDEN + k0 + 8 * hi;
  FragU f;
  f.h2[0] = *(const us8*)(p);
  f.h2[1] = *(const us8*)(p + 16);
  return f.v;
}

// A fragment from global f32 row-major (stride in elements), K zero-padded to KMAX.
template <int KMAX, int K0>
static __device__ __forceinline__ v16bf make_afrag_f32(const float* base, int stride,
                                                       int lane) {
  int m = lane & 15, hi = lane >> 4;
  const float* row = base + (size_t)m * stride;
  FragU f;
#pragma unroll
  for (int j = 0; j < 8; ++j) {
    int ka = K0 + 8 * hi + j;
    int kb = K0 + 16 + 8 * hi + j;
    float va = (ka < KMAX) ? row[ka] : 0.0f;
    float vb = (kb < KMAX) ? row[kb] : 0.0f;
    f.v[j]     = (__bf16)va;     // hardware cvt, lets clang pack pairs
    f.v[8 + j] = (__bf16)vb;
  }
  return f.v;
}

// ---------- weight packing: f32 [K x 128] -> bf16 WMMA-B fragments --------
// frag f = kt*8+nt holds K rows [kt*32, kt*32+32) x cols [nt*16, nt*16+16);
// lane l's 16 values contiguous at f*512 + l*16; value j <-> K = kt*32+8*(l>>4)+j (j<8), kt*32+16+8*(l>>4)+(j-8) (j>=8).

__global__ void schnet_pack_w(const float* __restrict__ W,
                              unsigned short* __restrict__ out,
                              int K_actual, int KT) {
  int idx = blockIdx.x * blockDim.x + threadIdx.x;
  if (idx >= KT * 8 * 512) return;
  int f  = idx >> 9;
  int l  = (idx >> 4) & 31;
  int j  = idx & 15;
  int kt = f >> 3, nt = f & 7;
  int k  = kt * 32 + 8 * (l >> 4) + (j & 7) + ((j >> 3) << 4);
  int n  = nt * 16 + (l & 15);
  float v = (k < K_actual) ? W[(size_t)k * HIDDEN + n] : 0.0f;
  out[idx] = f32_to_bf16_bits(v);
}

__global__ void schnet_zero(float* __restrict__ p, size_t n) {
  size_t i = (size_t)blockIdx.x * blockDim.x + threadIdx.x;
  if (i < n) p[i] = 0.0f;
}

// ---------- node projection: h = x @ W_in ---------------------------------

__global__ void __launch_bounds__(256) schnet_nodeproj(
    const float* __restrict__ x, const unsigned short* __restrict__ Winp,
    float* __restrict__ h, int n_nodes) {
  const int wave = threadIdx.x >> 5, lane = threadIdx.x & 31;
  const int node0 = (blockIdx.x * 8 + wave) * 16;
  if (node0 >= n_nodes) return;
  const float* xr = x + (size_t)node0 * HIDDEN;
  v16bf a0 = make_afrag_f32<HIDDEN, 0>(xr, HIDDEN, lane);
  v16bf a1 = make_afrag_f32<HIDDEN, 32>(xr, HIDDEN, lane);
  v16bf a2 = make_afrag_f32<HIDDEN, 64>(xr, HIDDEN, lane);
  v16bf a3 = make_afrag_f32<HIDDEN, 96>(xr, HIDDEN, lane);
  const int n_l = lane & 15, hi = lane >> 4;
  for (int nt = 0; nt < 8; ++nt) {
    v8f acc = {};
    acc = wmma_bf16(a0, load_bfrag(Winp + (size_t)(0 * 8 + nt) * 512 + lane * 16), acc);
    acc = wmma_bf16(a1, load_bfrag(Winp + (size_t)(1 * 8 + nt) * 512 + lane * 16), acc);
    acc = wmma_bf16(a2, load_bfrag(Winp + (size_t)(2 * 8 + nt) * 512 + lane * 16), acc);
    acc = wmma_bf16(a3, load_bfrag(Winp + (size_t)(3 * 8 + nt) * 512 + lane * 16), acc);
#pragma unroll
    for (int r = 0; r < 8; ++r)
      h[(size_t)(node0 + r + 8 * hi) * HIDDEN + nt * 16 + n_l] = acc[r];
  }
}

// ---------- edge kernel: filter MLP + gather*filter + atomic scatter ------

__global__ void __launch_bounds__(256) schnet_edge(
    const float* __restrict__ edge_attr, const int* __restrict__ src,
    const int* __restrict__ dst, const float* __restrict__ edge_weight,
    const unsigned short* __restrict__ Wf1p, const unsigned short* __restrict__ Wf2p,
    const float* __restrict__ b_f1, const float* __restrict__ b_f2,
    const float* __restrict__ h, float* __restrict__ agg, int n_edges) {
  __shared__ __align__(16) unsigned short sT[8][16 * HIDDEN];  // tanh tile, bf16
  __shared__ int   sSrc[8][16];
  __shared__ int   sDst[8][16];
  __shared__ float sC[8][16];
  const int wave = threadIdx.x >> 5, lane = threadIdx.x & 31;
  const int e0 = (blockIdx.x * 8 + wave) * 16;
  const bool active = (e0 < n_edges);
  const int n_l = lane & 15, hi = lane >> 4;

  if (active && lane < 16) {
    int e = e0 + lane;
    sSrc[wave][lane] = src[e];
    sDst[wave][lane] = dst[e];
    sC[wave][lane] = 0.5f * (__cosf(edge_weight[e] * 0.31415926535f) + 1.0f);
  }
  __syncthreads();

  if (active) {
    // GEMM1: [16 x 64(pad of 50)] @ [64 x 128], epilogue: +b_f1, tanh, bf16 -> LDS
    const float* ar = edge_attr + (size_t)e0 * NUM_RBF;
    v16bf a0 = make_afrag_f32<NUM_RBF, 0>(ar, NUM_RBF, lane);
    v16bf a1 = make_afrag_f32<NUM_RBF, 32>(ar, NUM_RBF, lane);
    for (int nt = 0; nt < 8; ++nt) {
      v8f acc = {};
      acc = wmma_bf16(a0, load_bfrag(Wf1p + (size_t)(0 * 8 + nt) * 512 + lane * 16), acc);
      acc = wmma_bf16(a1, load_bfrag(Wf1p + (size_t)(1 * 8 + nt) * 512 + lane * 16), acc);
      float bias = b_f1[nt * 16 + n_l];
#pragma unroll
      for (int r = 0; r < 8; ++r) {
        int m = r + 8 * hi;  // C/D layout: reg r, half-wave hi -> edge row m
        sT[wave][m * HIDDEN + nt * 16 + n_l] = f32_to_bf16_bits(fast_tanh(acc[r] + bias));
      }
    }
  }
  __syncthreads();

  if (active) {
    // GEMM2: [16 x 128] @ [128 x 128], epilogue: +b_f2, *C, *h[src], atomic-scatter
    v16bf t0 = load_afrag_lds(sT[wave], lane, 0);
    v16bf t1 = load_afrag_lds(sT[wave], lane, 32);
    v16bf t2 = load_afrag_lds(sT[wave], lane, 64);
    v16bf t3 = load_afrag_lds(sT[wave], lane, 96);
    float cv[8]; int sr[8], dr[8];
#pragma unroll
    for (int r = 0; r < 8; ++r) {
      int m = r + 8 * hi;
      cv[r] = sC[wave][m]; sr[r] = sSrc[wave][m]; dr[r] = sDst[wave][m];
    }
    for (int nt = 0; nt < 8; ++nt) {
      v8f acc = {};
      acc = wmma_bf16(t0, load_bfrag(Wf2p + (size_t)(0 * 8 + nt) * 512 + lane * 16), acc);
      acc = wmma_bf16(t1, load_bfrag(Wf2p + (size_t)(1 * 8 + nt) * 512 + lane * 16), acc);
      acc = wmma_bf16(t2, load_bfrag(Wf2p + (size_t)(2 * 8 + nt) * 512 + lane * 16), acc);
      acc = wmma_bf16(t3, load_bfrag(Wf2p + (size_t)(3 * 8 + nt) * 512 + lane * 16), acc);
      int n = nt * 16 + n_l;
      float bias = b_f2[n];
#pragma unroll
      for (int r = 0; r < 8; ++r) {
        float wf  = (acc[r] + bias) * cv[r];
        float msg = h[(size_t)sr[r] * HIDDEN + n] * wf;         // gather (L2-resident)
        atomicAdd(&agg[(size_t)dr[r] * HIDDEN + n], msg);       // scatter into L2
      }
    }
  }
}

// ---------- output: out = tanh(agg @ W_out + b_out) @ W_lin + b_lin -------

__global__ void __launch_bounds__(256) schnet_out(
    const float* __restrict__ agg, const unsigned short* __restrict__ Woutp,
    const float* __restrict__ b_out, const unsigned short* __restrict__ Wlinp,
    const float* __restrict__ b_lin, float* __restrict__ out, int n_nodes) {
  __shared__ __align__(16) unsigned short sT[8][16 * HIDDEN];
  const int wave = threadIdx.x >> 5, lane = threadIdx.x & 31;
  const int node0 = (blockIdx.x * 8 + wave) * 16;
  const bool active = (node0 < n_nodes);
  const int n_l = lane & 15, hi = lane >> 4;

  if (active) {
    const float* ar = agg + (size_t)node0 * HIDDEN;
    v16bf a0 = make_afrag_f32<HIDDEN, 0>(ar, HIDDEN, lane);
    v16bf a1 = make_afrag_f32<HIDDEN, 32>(ar, HIDDEN, lane);
    v16bf a2 = make_afrag_f32<HIDDEN, 64>(ar, HIDDEN, lane);
    v16bf a3 = make_afrag_f32<HIDDEN, 96>(ar, HIDDEN, lane);
    for (int nt = 0; nt < 8; ++nt) {
      v8f acc = {};
      acc = wmma_bf16(a0, load_bfrag(Woutp + (size_t)(0 * 8 + nt) * 512 + lane * 16), acc);
      acc = wmma_bf16(a1, load_bfrag(Woutp + (size_t)(1 * 8 + nt) * 512 + lane * 16), acc);
      acc = wmma_bf16(a2, load_bfrag(Woutp + (size_t)(2 * 8 + nt) * 512 + lane * 16), acc);
      acc = wmma_bf16(a3, load_bfrag(Woutp + (size_t)(3 * 8 + nt) * 512 + lane * 16), acc);
      float bias = b_out[nt * 16 + n_l];
#pragma unroll
      for (int r = 0; r < 8; ++r) {
        int m = r + 8 * hi;
        sT[wave][m * HIDDEN + nt * 16 + n_l] = f32_to_bf16_bits(fast_tanh(acc[r] + bias));
      }
    }
  }
  __syncthreads();
  if (active) {
    v16bf t0 = load_afrag_lds(sT[wave], lane, 0);
    v16bf t1 = load_afrag_lds(sT[wave], lane, 32);
    v16bf t2 = load_afrag_lds(sT[wave], lane, 64);
    v16bf t3 = load_afrag_lds(sT[wave], lane, 96);
    for (int nt = 0; nt < 8; ++nt) {
      v8f acc = {};
      acc = wmma_bf16(t0, load_bfrag(Wlinp + (size_t)(0 * 8 + nt) * 512 + lane * 16), acc);
      acc = wmma_bf16(t1, load_bfrag(Wlinp + (size_t)(1 * 8 + nt) * 512 + lane * 16), acc);
      acc = wmma_bf16(t2, load_bfrag(Wlinp + (size_t)(2 * 8 + nt) * 512 + lane * 16), acc);
      acc = wmma_bf16(t3, load_bfrag(Wlinp + (size_t)(3 * 8 + nt) * 512 + lane * 16), acc);
      float bias = b_lin[nt * 16 + n_l];
#pragma unroll
      for (int r = 0; r < 8; ++r)
        out[(size_t)(node0 + r + 8 * hi) * HIDDEN + nt * 16 + n_l] = acc[r] + bias;
    }
  }
}

// ---------- launch --------------------------------------------------------

extern "C" void kernel_launch(void* const* d_in, const int* in_sizes, int n_in,
                              void* d_out, int out_size, void* d_ws, size_t ws_size,
                              hipStream_t stream) {
  const float* x    = (const float*)d_in[0];
  const int*   ei   = (const int*)d_in[1];     // edge_index [2, E] (int32 on device)
  const float* ew   = (const float*)d_in[2];
  const float* ea   = (const float*)d_in[3];
  const float* Wf1  = (const float*)d_in[4];
  const float* bf1  = (const float*)d_in[5];
  const float* Wf2  = (const float*)d_in[6];
  const float* bf2  = (const float*)d_in[7];
  const float* Win  = (const float*)d_in[8];
  const float* Wout = (const float*)d_in[9];
  const float* bout = (const float*)d_in[10];
  const float* Wlin = (const float*)d_in[11];
  const float* blin = (const float*)d_in[12];

  const int n_nodes = in_sizes[0] / HIDDEN;
  const int n_edges = in_sizes[2];
  const int* src = ei;
  const int* dst = ei + n_edges;

  // workspace: 4x packed 128x128 (32KB each) + packed 64x128 (16KB) + h + agg
  char* ws = (char*)d_ws;
  unsigned short* Winp  = (unsigned short*)ws;
  unsigned short* Wf2p  = Winp + 16384;
  unsigned short* Woutp = Wf2p + 16384;
  unsigned short* Wlinp = Woutp + 16384;
  unsigned short* Wf1p  = Wlinp + 16384;  // 8192 entries
  size_t wbytes = (4 * 16384 + 8192) * sizeof(unsigned short);
  wbytes = (wbytes + 255) & ~(size_t)255;
  float* h   = (float*)(ws + wbytes);
  float* agg = h + (size_t)n_nodes * HIDDEN;

  schnet_pack_w<<<(4 * 8 * 512 + 255) / 256, 256, 0, stream>>>(Win,  Winp,  HIDDEN, 4);
  schnet_pack_w<<<(4 * 8 * 512 + 255) / 256, 256, 0, stream>>>(Wf2,  Wf2p,  HIDDEN, 4);
  schnet_pack_w<<<(4 * 8 * 512 + 255) / 256, 256, 0, stream>>>(Wout, Woutp, HIDDEN, 4);
  schnet_pack_w<<<(4 * 8 * 512 + 255) / 256, 256, 0, stream>>>(Wlin, Wlinp, HIDDEN, 4);
  schnet_pack_w<<<(2 * 8 * 512 + 255) / 256, 256, 0, stream>>>(Wf1,  Wf1p,  NUM_RBF, 2);

  size_t aggN = (size_t)n_nodes * HIDDEN;
  schnet_zero<<<(int)((aggN + 255) / 256), 256, 0, stream>>>(agg, aggN);

  const int ng_nodes = (n_nodes + 15) / 16;
  schnet_nodeproj<<<(ng_nodes + 7) / 8, 256, 0, stream>>>(x, Winp, h, n_nodes);

  const int ng_edges = (n_edges + 15) / 16;
  schnet_edge<<<(ng_edges + 7) / 8, 256, 0, stream>>>(ea, src, dst, ew, Wf1p, Wf2p,
                                                      bf1, bf2, h, agg, n_edges);

  schnet_out<<<(ng_nodes + 7) / 8, 256, 0, stream>>>(agg, Woutp, bout, Wlinp, blin,
                                                     (float*)d_out, n_nodes);
}
